// SparseSelfAttention_9990093930821
// MI455X (gfx1250) — matrix operand order repaired and checked by reading, not compiled, not run on hardware
//
#include <hip/hip_runtime.h>
#include <hip/hip_bf16.h>

// ---------------------------------------------------------------------------
// Sparse causal self-attention forward for MI455X (gfx1250, wave32, WMMA).
// B=1, T=2048, C=768, NH=12, HD=64.
// All matrix math: v_wmma_f32_16x16x32_bf16 (f32 accumulate).
//  - every WMMA fragment is 2x b128 per lane (weights pre-transposed, V
//    written transposed, LDS P-tile padded to 16B-aligned stride)
//  - fragment loads are software-pipelined (double-buffered) so WMMAs only
//    wait on loads issued one iteration earlier
// Attention is flash-style (online softmax); adj (201 MB) is the only
// T*T-sized HBM traffic => ~9 us floor at 23.3 TB/s.
// ---------------------------------------------------------------------------

#define TT 2048
#define CC 768
#define NHH 12
#define HDD 64
#define C3 2304

typedef __bf16 bf16;
typedef __attribute__((ext_vector_type(16))) __bf16 v16bf;
typedef __attribute__((ext_vector_type(8)))  __bf16 v8bf;
typedef __attribute__((ext_vector_type(8)))  float  v8f;

__device__ __forceinline__ bf16 f2bf(float f) {
  unsigned u = __builtin_bit_cast(unsigned, f);
  u += 0x7FFFu + ((u >> 16) & 1u);            // round-to-nearest-even
  unsigned short h = (unsigned short)(u >> 16);
  return __builtin_bit_cast(bf16, h);
}

// ---- WMMA fragment loaders (contiguous k order, same in A and B) -----
// All loaders are 2x 16-byte loads per lane.

// A fragment: 16x32 bf16 from row-major tile (ld in elements, multiple of 8).
// lane L holds row M=L&15; element e holds k = e + 16*(L>>4).
__device__ __forceinline__ v16bf load_a(const bf16* __restrict__ src, int ld, int lane) {
  const bf16* r = src + (size_t)(lane & 15) * ld + 16 * (lane >> 4);
  v8bf lo = *(const v8bf*)(r);       // k = base+0..7
  v8bf hi = *(const v8bf*)(r + 8);   // k = base+8..15
  return __builtin_shufflevector(lo, hi, 0, 1, 2, 3, 4, 5, 6, 7,
                                         8, 9, 10, 11, 12, 13, 14, 15);
}

// B fragment from a TRANSPOSED source: B[k][n] = src[n*ld + k].
// lane L holds col N=L&15; element e holds k = e + 16*(L>>4) -> 16 contiguous.
__device__ __forceinline__ v16bf load_bT(const bf16* __restrict__ src, int ld, int lane) {
  const bf16* r = src + (size_t)(lane & 15) * ld + 16 * (lane >> 4);
  v8bf lo = *(const v8bf*)(r);
  v8bf hi = *(const v8bf*)(r + 8);
  return __builtin_shufflevector(lo, hi, 0, 1, 2, 3, 4, 5, 6, 7,
                                         8, 9, 10, 11, 12, 13, 14, 15);
}

__device__ __forceinline__ v8f wmma_bf16(v16bf a, v16bf b, v8f c) {
  return __builtin_amdgcn_wmma_f32_16x16x32_bf16(false, a, false, b, (short)0, c,
                                                 false, false);
}

// ---------------------------------------------------------------------------
// f32 -> bf16, 8 elements per thread (2x float4 in, one 16B store out).
__global__ void cvt_f32_bf16(const float* __restrict__ in, bf16* __restrict__ out,
                             int n8) {
  int i = blockIdx.x * blockDim.x + threadIdx.x;
  int st = gridDim.x * blockDim.x;
  for (; i < n8; i += st) {
    const float4* p = (const float4*)(in + (size_t)i * 8);
    float4 a = p[0], b = p[1];
    v8bf o;
    o[0] = f2bf(a.x); o[1] = f2bf(a.y); o[2] = f2bf(a.z); o[3] = f2bf(a.w);
    o[4] = f2bf(b.x); o[5] = f2bf(b.y); o[6] = f2bf(b.z); o[7] = f2bf(b.w);
    *(v8bf*)(out + (size_t)i * 8) = o;
  }
}

// f32 [K][N] -> bf16 transposed [N][K]; writes coalesced, reads L2-resident.
__global__ void cvt_transpose_bf16(const float* __restrict__ in, bf16* __restrict__ out,
                                   int Kdim, int Ndim) {
  int i = blockIdx.x * blockDim.x + threadIdx.x;
  int st = gridDim.x * blockDim.x;
  int total = Kdim * Ndim;
  for (; i < total; i += st) {
    int j = i / Kdim;                  // output row  = N index
    int k = i - j * Kdim;              // output col  = K index
    out[i] = f2bf(in[(size_t)k * Ndim + j]);
  }
}

__global__ void zero_f32x4(float* __restrict__ p, int n4) {
  int i = blockIdx.x * blockDim.x + threadIdx.x;
  int st = gridDim.x * blockDim.x;
  float4* p4 = (float4*)p;
  float4 z = make_float4(0.f, 0.f, 0.f, 0.f);
  for (; i < n4; i += st) p4[i] = z;
}

// ---------------------------------------------------------------------------
// qkv = x @ W_attn + b_attn.  W supplied transposed (WT[j][k], bf16).
// One wave computes a 16(M) x 64(N) strip; software-pipelined K loop.
// Epilogue scatters to Q,K [NH][T][HD] and V transposed [NH][HD][T].
__global__ void __launch_bounds__(256) gemm_qkv(
    const bf16* __restrict__ X, const bf16* __restrict__ WT,
    const float* __restrict__ bias,
    bf16* __restrict__ Q, bf16* __restrict__ K, bf16* __restrict__ Vt) {
  int lane = threadIdx.x & 31;
  int wv   = blockIdx.x * 8 + (threadIdx.x >> 5);   // 128 * 36 waves total
  int tm = wv / (C3 / 64);
  int tn = wv % (C3 / 64);
  int m0 = tm * 16, n0 = tn * 64;

  const bf16* Xp = X  + (size_t)m0 * CC;
  const bf16* W0 = WT + (size_t)(n0 +  0) * CC;
  const bf16* W1 = WT + (size_t)(n0 + 16) * CC;
  const bf16* W2 = WT + (size_t)(n0 + 32) * CC;
  const bf16* W3 = WT + (size_t)(n0 + 48) * CC;

  v8f acc0 = {}, acc1 = {}, acc2 = {}, acc3 = {};
  // prologue loads
  v16bf a  = load_a (Xp, CC, lane);
  v16bf b0 = load_bT(W0, CC, lane);
  v16bf b1 = load_bT(W1, CC, lane);
  v16bf b2 = load_bT(W2, CC, lane);
  v16bf b3 = load_bT(W3, CC, lane);
  for (int k0 = 32; k0 < CC; k0 += 32) {
    // issue next-iteration loads first -> WMMAs below overlap them
    v16bf an  = load_a (Xp + k0, CC, lane);
    v16bf bn0 = load_bT(W0 + k0, CC, lane);
    v16bf bn1 = load_bT(W1 + k0, CC, lane);
    v16bf bn2 = load_bT(W2 + k0, CC, lane);
    v16bf bn3 = load_bT(W3 + k0, CC, lane);
    acc0 = wmma_bf16(a, b0, acc0);
    acc1 = wmma_bf16(a, b1, acc1);
    acc2 = wmma_bf16(a, b2, acc2);
    acc3 = wmma_bf16(a, b3, acc3);
    a = an; b0 = bn0; b1 = bn1; b2 = bn2; b3 = bn3;
  }
  acc0 = wmma_bf16(a, b0, acc0);                    // drain
  acc1 = wmma_bf16(a, b1, acc1);
  acc2 = wmma_bf16(a, b2, acc2);
  acc3 = wmma_bf16(a, b3, acc3);

  int half = lane >> 4, col = lane & 15;
  int which = n0 / CC;                  // uniform over the 64-wide strip
  int h = (n0 % CC) >> 6;               // uniform over the 64-wide strip
  v8f* accs[4] = {&acc0, &acc1, &acc2, &acc3};
#pragma unroll
  for (int t = 0; t < 4; ++t) {
    v8f av = *accs[t];
    int d = t * 16 + col;               // 0..63 within the head
    float bv = bias[n0 + t * 16 + col];
    bf16* base;
    size_t stride;
    if (which == 0)      { base = Q  + ((size_t)h * TT) * HDD + d; stride = HDD; }
    else if (which == 1) { base = K  + ((size_t)h * TT) * HDD + d; stride = HDD; }
    else                 { base = Vt + ((size_t)h * HDD + d) * TT; stride = 1;   }
#pragma unroll
    for (int r = 0; r < 8; ++r) {
      int m = m0 + r + 8 * half;
      base[(size_t)m * stride] = f2bf(av[r] + bv);
    }
  }
}

// ---------------------------------------------------------------------------
// Flash attention with hard gate + adj output.
// Block = 4 waves = 64 query rows of one head; wave w owns rows q0..q0+15.
// Pipeline per key block: S-WMMAs use K-frags loaded last iteration; next
// K-frags and current V-frags are issued before the softmax VALU section.
// Uniform trip count per block -> __syncthreads is safe.
__global__ void __launch_bounds__(128) attn_kernel(
    const bf16* __restrict__ Q, const bf16* __restrict__ K,
    const bf16* __restrict__ Vt, bf16* __restrict__ Y,
    float* __restrict__ adj) {
  // stride 40 elems = 80 B: 16B-aligned per-row fragments, conflict-free banks
  __shared__ bf16 Plds[4][16][40];

  int lane = threadIdx.x & 31;
  int wave = threadIdx.x >> 5;
  int h  = blockIdx.x / (TT / 64);
  int qb = blockIdx.x % (TT / 64);
  int q0 = qb * 64 + wave * 16;

  const bf16* Qh  = Q  + (size_t)h * TT * HDD;   // [T][HD]
  const bf16* Kh  = K  + (size_t)h * TT * HDD;   // [T][HD]
  const bf16* Vth = Vt + (size_t)h * HDD * TT;   // [HD][T]
  float* adjh = adj + (size_t)h * TT * TT;

  // Q fragments are loop-invariant (HD=64 -> two 16x32 A fragments).
  v16bf aq0 = load_a(Qh + (size_t)q0 * HDD + 0,  HDD, lane);
  v16bf aq1 = load_a(Qh + (size_t)q0 * HDD + 32, HDD, lane);

  v8f y0 = {}, y1 = {}, y2 = {}, y3 = {};
  float rm[8], rz[8];
#pragma unroll
  for (int r = 0; r < 8; ++r) { rm[r] = -__builtin_inff(); rz[r] = 0.f; }

  int half = lane >> 4, col = lane & 15;
  int kend = qb * 64 + 64;                       // causal: no keys beyond block

  // prologue: K fragments for the first key block
  v16bf kb00 = load_bT(Kh + (size_t)0 * HDD +  0, HDD, lane);
  v16bf kb01 = load_bT(Kh + (size_t)0 * HDD + 32, HDD, lane);
  v16bf kb10 = load_bT(Kh + (size_t)16 * HDD +  0, HDD, lane);
  v16bf kb11 = load_bT(Kh + (size_t)16 * HDD + 32, HDD, lane);

  for (int k0 = 0; k0 < kend; k0 += 32) {
    // S = Q K^T : two 16x16 N-tiles, contraction HD=64 in two k-steps.
    v8f s0 = {}, s1 = {};
    s0 = wmma_bf16(aq0, kb00, s0);
    s0 = wmma_bf16(aq1, kb01, s0);
    s1 = wmma_bf16(aq0, kb10, s1);
    s1 = wmma_bf16(aq1, kb11, s1);

    // prefetch next key block's K fragments (uniform branch)
    if (k0 + 32 < kend) {
      kb00 = load_bT(Kh + (size_t)(k0 + 32) * HDD +  0, HDD, lane);
      kb01 = load_bT(Kh + (size_t)(k0 + 32) * HDD + 32, HDD, lane);
      kb10 = load_bT(Kh + (size_t)(k0 + 48) * HDD +  0, HDD, lane);
      kb11 = load_bT(Kh + (size_t)(k0 + 48) * HDD + 32, HDD, lane);
    }
    // V fragments for this key block (consumed after the softmax section).
    // Vt is [HD][T]: B[k][n] = Vt[d0+n][key0+k] -> transposed-source loader.
    v16bf vb0 = load_bT(Vth + (size_t)( 0) * TT + k0, TT, lane);
    v16bf vb1 = load_bT(Vth + (size_t)(16) * TT + k0, TT, lane);
    v16bf vb2 = load_bT(Vth + (size_t)(32) * TT + k0, TT, lane);
    v16bf vb3 = load_bT(Vth + (size_t)(48) * TT + k0, TT, lane);

#pragma unroll
    for (int r = 0; r < 8; ++r) {
      int row = q0 + r + 8 * half;
      int c0 = k0 + col, c1 = k0 + 16 + col;
      float sv0 = s0[r] * 0.125f;                // 1/sqrt(64)
      float sv1 = s1[r] * 0.125f;
      bool v0 = (c0 <= row), v1 = (c1 <= row);
      adjh[(size_t)row * TT + c0] = (v0 && (sv0 + 1.f > 0.f)) ? 1.f : 0.f;
      adjh[(size_t)row * TT + c1] = (v1 && (sv1 + 1.f > 0.f)) ? 1.f : 0.f;
      if (!v0) sv0 = -__builtin_inff();
      if (!v1) sv1 = -__builtin_inff();

      // row max over the 32-key block: reduce across the 16 lanes of this half
      float bm = fmaxf(sv0, sv1);
      bm = fmaxf(bm, __shfl_xor(bm, 1, 32));
      bm = fmaxf(bm, __shfl_xor(bm, 2, 32));
      bm = fmaxf(bm, __shfl_xor(bm, 4, 32));
      bm = fmaxf(bm, __shfl_xor(bm, 8, 32));
      float mnew = fmaxf(rm[r], bm);             // finite after first block
      float fs = __expf(rm[r] - mnew);

      float p0 = __expf(sv0 - mnew);             // masked -> 0
      float p1 = __expf(sv1 - mnew);
      float ps = p0 + p1;
      ps += __shfl_xor(ps, 1, 32);
      ps += __shfl_xor(ps, 2, 32);
      ps += __shfl_xor(ps, 4, 32);
      ps += __shfl_xor(ps, 8, 32);
      rz[r] = rz[r] * fs + ps;                   // softmax denominator (ungated)
      rm[r] = mnew;
      y0[r] *= fs; y1[r] *= fs; y2[r] *= fs; y3[r] *= fs;

      // hard gate multiplies the numerator only
      float g0 = (sv0 + 1.f > 0.f) ? p0 : 0.f;
      float g1 = (sv1 + 1.f > 0.f) ? p1 : 0.f;
      Plds[wave][r + 8 * half][col]      = f2bf(g0);
      Plds[wave][r + 8 * half][col + 16] = f2bf(g1);
    }
    __syncthreads();

    // P (16x32, C-layout) -> A fragment via LDS, then y += P @ V.
    v16bf pa = load_a(&Plds[wave][0][0], 40, lane);
    y0 = wmma_bf16(pa, vb0, y0);
    y1 = wmma_bf16(pa, vb1, y1);
    y2 = wmma_bf16(pa, vb2, y2);
    y3 = wmma_bf16(pa, vb3, y3);
    __syncthreads();
  }

  // epilogue: divide by softmax sum, merge heads into [T][C] bf16
#pragma unroll
  for (int r = 0; r < 8; ++r) {
    int row = q0 + r + 8 * half;
    float inv = 1.f / rz[r];                     // diagonal present -> rz > 0
    bf16* yr = Y + (size_t)row * CC + h * HDD;
    yr[ 0 + col] = f2bf(y0[r] * inv);
    yr[16 + col] = f2bf(y1[r] * inv);
    yr[32 + col] = f2bf(y2[r] * inv);
    yr[48 + col] = f2bf(y3[r] * inv);
  }
}

// ---------------------------------------------------------------------------
// out = Yattn @ W_proj + b_proj.  W supplied transposed.  16x64 per wave,
// software-pipelined K loop.
__global__ void __launch_bounds__(256) gemm_proj(
    const bf16* __restrict__ A, const bf16* __restrict__ WT,
    const float* __restrict__ bias, float* __restrict__ out) {
  int lane = threadIdx.x & 31;
  int wv   = blockIdx.x * 8 + (threadIdx.x >> 5);   // 128 * 12 waves total
  int tm = wv / (CC / 64);
  int tn = wv % (CC / 64);
  int m0 = tm * 16, n0 = tn * 64;

  const bf16* Ap = A  + (size_t)m0 * CC;
  const bf16* W0 = WT + (size_t)(n0 +  0) * CC;
  const bf16* W1 = WT + (size_t)(n0 + 16) * CC;
  const bf16* W2 = WT + (size_t)(n0 + 32) * CC;
  const bf16* W3 = WT + (size_t)(n0 + 48) * CC;

  v8f acc0 = {}, acc1 = {}, acc2 = {}, acc3 = {};
  v16bf a  = load_a (Ap, CC, lane);
  v16bf b0 = load_bT(W0, CC, lane);
  v16bf b1 = load_bT(W1, CC, lane);
  v16bf b2 = load_bT(W2, CC, lane);
  v16bf b3 = load_bT(W3, CC, lane);
  for (int k0 = 32; k0 < CC; k0 += 32) {
    v16bf an  = load_a (Ap + k0, CC, lane);
    v16bf bn0 = load_bT(W0 + k0, CC, lane);
    v16bf bn1 = load_bT(W1 + k0, CC, lane);
    v16bf bn2 = load_bT(W2 + k0, CC, lane);
    v16bf bn3 = load_bT(W3 + k0, CC, lane);
    acc0 = wmma_bf16(a, b0, acc0);
    acc1 = wmma_bf16(a, b1, acc1);
    acc2 = wmma_bf16(a, b2, acc2);
    acc3 = wmma_bf16(a, b3, acc3);
    a = an; b0 = bn0; b1 = bn1; b2 = bn2; b3 = bn3;
  }
  acc0 = wmma_bf16(a, b0, acc0);
  acc1 = wmma_bf16(a, b1, acc1);
  acc2 = wmma_bf16(a, b2, acc2);
  acc3 = wmma_bf16(a, b3, acc3);

  int half = lane >> 4, col = lane & 15;
  v8f* accs[4] = {&acc0, &acc1, &acc2, &acc3};
#pragma unroll
  for (int t = 0; t < 4; ++t) {
    v8f av = *accs[t];
    float bv = bias[n0 + t * 16 + col];
#pragma unroll
    for (int r = 0; r < 8; ++r) {
      int m = m0 + r + 8 * half;
      out[(size_t)m * CC + n0 + t * 16 + col] = av[r] + bv;
    }
  }
}

// ---------------------------------------------------------------------------
extern "C" void kernel_launch(void* const* d_in, const int* in_sizes, int n_in,
                              void* d_out, int out_size, void* d_ws, size_t ws_size,
                              hipStream_t stream) {
  (void)in_sizes; (void)n_in; (void)out_size; (void)ws_size;
  const float* x      = (const float*)d_in[0];
  const float* W_attn = (const float*)d_in[1];
  const float* b_attn = (const float*)d_in[2];
  const float* W_proj = (const float*)d_in[3];
  const float* b_proj = (const float*)d_in[4];

  char* ws = (char*)d_ws;
  size_t off = 0;
  auto alloc = [&](size_t elems) {
    bf16* p = (bf16*)(ws + off);
    off += ((elems * sizeof(bf16) + 255) / 256) * 256;
    return p;
  };
  bf16* xb   = alloc((size_t)TT * CC);        // x in bf16
  bf16* watT = alloc((size_t)C3 * CC);        // W_attn^T bf16  [2304][768]
  bf16* wprT = alloc((size_t)CC * CC);        // W_proj^T bf16  [768][768]
  bf16* Qb   = alloc((size_t)NHH * TT * HDD); // Q [NH][T][HD]
  bf16* Kb   = alloc((size_t)NHH * TT * HDD); // K [NH][T][HD]
  bf16* Vtb  = alloc((size_t)NHH * HDD * TT); // V^T [NH][HD][T]
  bf16* Yb   = alloc((size_t)TT * CC);        // merged-head attention output

  float* y_out   = (float*)d_out;                   // [T][C] f32
  float* adj_out = y_out + (size_t)TT * CC;         // [NH][T][T] as 0/1 f32

  cvt_f32_bf16<<<512, 256, 0, stream>>>(x, xb, (TT * CC) / 8);
  cvt_transpose_bf16<<<1024, 256, 0, stream>>>(W_attn, watT, CC, C3);
  cvt_transpose_bf16<<<1024, 256, 0, stream>>>(W_proj, wprT, CC, CC);

  // upper-triangular key blocks are never visited by attn_kernel -> pre-zero
  zero_f32x4<<<2048, 256, 0, stream>>>(adj_out, (NHH * TT * TT) / 4);

  gemm_qkv<<<(128 * (C3 / 64)) / 8, 256, 0, stream>>>(xb, watT, b_attn, Qb, Kb, Vtb);
  attn_kernel<<<NHH * (TT / 64), 128, 0, stream>>>(Qb, Kb, Vtb, Yb, adj_out);
  gemm_proj<<<(128 * (CC / 64)) / 8, 256, 0, stream>>>(Yb, wprT, b_proj, y_out);
}
